// GraphConv_37675453120712
// MI455X (gfx1250) — matrix operand order repaired and checked by reading, not compiled, not run on hardware
//
#include <hip/hip_runtime.h>
#include <hip/hip_bf16.h>

// GCN layer on MI455X (gfx1250):
//   out = fxp( D_row^{-1/2} * (A^T @ (D_col^{-1/2} * fxp(x@w))) + b )
// Pipeline: K_deg (degrees, 1 GiB read) -> K_norm (reduce) ->
//           K_xw (y=fxp(x@w), colsum scale, fp16 hi/lo split, transposed store) ->
//           K_agg (A^T @ Z via v_wmma_f32_16x16x32_f16, hi+lo accumulated).

typedef __attribute__((ext_vector_type(16))) _Float16 v16h;
typedef __attribute__((ext_vector_type(8)))  _Float16 v8h;
typedef __attribute__((ext_vector_type(8)))  float    v8f;
typedef __attribute__((ext_vector_type(4)))  float    v4f;

#define NN   16384
#define DIN  128
#define DOUT 64
#define FXP_S 65536.0f
#define FXP_R (1.0f/65536.0f)

static __device__ __forceinline__ float fxp_round(float v) {
    return __builtin_rintf(v * FXP_S) * FXP_R;   // round-to-nearest-even, /2^16 exact
}

static __device__ __forceinline__ v16h cat16(v8h a, v8h b) {
    return __builtin_shufflevector(a, b, 0,1,2,3,4,5,6,7,8,9,10,11,12,13,14,15);
}

static __device__ __forceinline__ v8f wmma_f16(v16h a, v16h b, v8f c) {
    return __builtin_amdgcn_wmma_f32_16x16x32_f16(false, a, false, b, (short)0, c,
                                                  false, false);
}

// ---------------------------------------------------------------------------
// K_deg: block b owns rows [64b, 64b+64) x all 16384 cols.
// Row sums accumulate privately (thread owns a fixed row); column sums via an
// LDS-staged 64x64 tile read column-wise into an LDS colsum array, with one
// per-block partial row written to colpartG. All summed values are 0/1 -> exact.
__global__ __launch_bounds__(256)
void gcn_deg(const float* __restrict__ A, float* __restrict__ colpartG,
             float* __restrict__ rowsumG) {
    extern __shared__ float sm[];
    float* colp = sm;                  // [16384]
    float* tile = sm + 16384;          // [64][68]
    float* rowp = sm + 16384 + 64*68;  // [64]
    const int t  = threadIdx.x;
    const int b  = blockIdx.x;
    const int r0 = b * 64;

    for (int i = t; i < NN; i += 256) colp[i] = 0.0f;
    if (t < 64) rowp[t] = 0.0f;
    __syncthreads();

    const int pr  = t >> 2;         // 0..63 : row within block
    const int pcb = (t & 3) * 16;   // col sub-base
    float rowacc = 0.0f;

    for (int c0 = 0; c0 < NN; c0 += 64) {
        const float* src = A + (size_t)(r0 + pr) * NN + c0 + pcb;
#pragma unroll
        for (int mm = 0; mm < 4; ++mm) {
            v4f v = *(const v4f*)(src + mm * 4);
            rowacc += v[0] + v[1] + v[2] + v[3];
            *(v4f*)&tile[pr * 68 + pcb + mm * 4] = v;
        }
        __syncthreads();
        // column phase: thread owns one column, sums 16 rows of the tile
        const int cl = t & 63;
        const int q0 = (t >> 6) * 16;
        float s = 0.0f;
#pragma unroll
        for (int q = 0; q < 16; ++q) s += tile[(q0 + q) * 68 + cl];
        atomicAdd(&colp[c0 + cl], s);
        __syncthreads();
    }

    atomicAdd(&rowp[pr], rowacc);
    __syncthreads();
    if (t < 64) rowsumG[r0 + t] = rowp[t];
    for (int i = t; i < NN; i += 256)
        colpartG[(size_t)b * NN + i] = colp[i];
}

// ---------------------------------------------------------------------------
// K_norm: reduce 256 column partials; emit isqrt(clip(deg,1)) for rows & cols.
__global__ __launch_bounds__(256)
void gcn_norm(const float* __restrict__ colpartG, const float* __restrict__ rowsumG,
              float* __restrict__ isc, float* __restrict__ isr) {
    const int tid = blockIdx.x * 256 + threadIdx.x;
    float s = 0.0f;
    for (int p = 0; p < 256; ++p) s += colpartG[(size_t)p * NN + tid];
    isc[tid] = rsqrtf(fmaxf(s, 1.0f));
    isr[tid] = rsqrtf(fmaxf(rowsumG[tid], 1.0f));
}

// ---------------------------------------------------------------------------
// K_xw: y = fxp(x @ w); z = isc[i]*y; split z into fp16 hi/lo; store transposed
// Zt[k][i] so K_agg's B-operand reads are contiguous. Block: 32 rows x 4 k-quads.
__global__ __launch_bounds__(128)
void gcn_xw(const float* __restrict__ x, const float* __restrict__ w,
            const float* __restrict__ isc,
            _Float16* __restrict__ Zh, _Float16* __restrict__ Zl) {
    __shared__ float xl[32 * 129];     // stride 129 -> conflict-free column reads
    __shared__ float wl[DIN * DOUT];
    const int t = threadIdx.x;

    for (int i = t; i < DIN * DOUT; i += 128) wl[i] = w[i];
    const size_t xbase = (size_t)blockIdx.x * 32 * DIN;
    for (int i = t; i < 32 * DIN; i += 128)
        xl[(i >> 7) * 129 + (i & 127)] = x[xbase + i];
    __syncthreads();

    const int il = t & 31;            // local row
    const int kb = t >> 5;            // 0..3 : owns k = kb*16 .. +16
    float acc[16];
#pragma unroll
    for (int kk = 0; kk < 16; ++kk) acc[kk] = 0.0f;

#pragma unroll 4
    for (int q = 0; q < DIN; ++q) {
        const float xv = xl[il * 129 + q];
#pragma unroll
        for (int g = 0; g < 4; ++g) {
            v4f wv = *(const v4f*)&wl[q * DOUT + kb * 16 + g * 4];
#pragma unroll
            for (int e = 0; e < 4; ++e) acc[g * 4 + e] += xv * wv[e];
        }
    }

    const int i = blockIdx.x * 32 + il;
    const float s = isc[i];
#pragma unroll
    for (int kk = 0; kk < 16; ++kk) {
        const float y  = fxp_round(acc[kk]);
        const float z  = s * y;
        const _Float16 zh = (_Float16)z;
        const _Float16 zl = (_Float16)(z - (float)zh);
        const int k = kb * 16 + kk;
        Zh[(size_t)k * NN + i] = zh;
        Zl[(size_t)k * NN + i] = zl;
    }
}

// ---------------------------------------------------------------------------
// K_agg: out^ = A^T @ Z. Block (128 thr = 4 waves) owns 64 output rows (j).
// Per K-step (32 i): stage A[32 x 64] fp32 -> fp16 transposed in LDS (exactly the
// WMMA A-operand layout: lane m = row j, contiguous K), stage Zt hi/lo slice,
// then 4 waves x 4 k-tiles x {hi,lo} = 8 v_wmma_f32_16x16x32_f16 per wave.
__global__ __launch_bounds__(128)
void gcn_agg(const float* __restrict__ A,
             const _Float16* __restrict__ Zh, const _Float16* __restrict__ Zl,
             const float* __restrict__ isr, const float* __restrict__ bias,
             float* __restrict__ out) {
    __shared__ _Float16 tileT[64 * 40];      // [j-local][i-local], stride 40 halves
    __shared__ _Float16 zlds[2 * 64 * 40];   // [hi/lo][k][i-local]

    const int t    = threadIdx.x;
    const int lane = t & 31;
    const int w    = t >> 5;                 // wave id -> j-tile
    const int m    = lane & 15;
    const int hi   = lane >> 4;
    const int jp   = blockIdx.x * 64;

    // staging index sets
    const int sr  = t >> 2;                  // 0..31 (i row)
    const int sc  = (t & 3) * 16;            // j sub-base
    const int zk  = t & 63;                  // k row for Zt staging
    const int zs  = t >> 6;                  // 0=hi, 1=lo

    v8f acc[4] = {};

    for (int i0 = 0; i0 < NN; i0 += 32) {
        // --- stage A chunk (transpose + fp32->fp16) ---
        const float* src = A + (size_t)(i0 + sr) * NN + jp + sc;
        if (i0 + 32 < NN) __builtin_prefetch(src + (size_t)32 * NN, 0, 1);
#pragma unroll
        for (int mm = 0; mm < 4; ++mm) {
            v4f v = *(const v4f*)(src + mm * 4);
#pragma unroll
            for (int e = 0; e < 4; ++e)
                tileT[(sc + mm * 4 + e) * 40 + sr] = (_Float16)v[e];
        }
        // --- stage Zt slice (hi & lo) ---
        const _Float16* zsrc = (zs ? Zl : Zh) + (size_t)zk * NN + i0;
#pragma unroll
        for (int mm = 0; mm < 4; ++mm)
            *(v8h*)&zlds[(zs * 64 + zk) * 40 + mm * 8] = *(const v8h*)(zsrc + mm * 8);
        __syncthreads();

        // --- WMMA: A-operand (ISA 16-bit A 16x32 layout) ---
        const int cb = (w * 16 + m) * 40;
        v8h a0 = *(const v8h*)&tileT[cb + hi * 8];        // K = hi*8 + 0..7
        v8h a1 = *(const v8h*)&tileT[cb + 16 + hi * 8];   // K = 16 + hi*8 + 0..7
        v16h Aop = cat16(a0, a1);

#pragma unroll
        for (int kt = 0; kt < 4; ++kt) {
            const int zb = (kt * 16 + m) * 40 + hi * 8;   // lane = col k, K = hi*8+0..7, 16+hi*8+0..7
            v8h bh0 = *(const v8h*)&zlds[zb];
            v8h bh1 = *(const v8h*)&zlds[zb + 16];
            acc[kt] = wmma_f16(Aop, cat16(bh0, bh1), acc[kt]);
            v8h bl0 = *(const v8h*)&zlds[64 * 40 + zb];
            v8h bl1 = *(const v8h*)&zlds[64 * 40 + zb + 16];
            acc[kt] = wmma_f16(Aop, cat16(bl0, bl1), acc[kt]);
        }
        __syncthreads();
    }

    // epilogue: C/D layout -> lane(m) = col, VGPR r -> row (r + 8*hi)
    float isrv[8];
#pragma unroll
    for (int r = 0; r < 8; ++r) isrv[r] = isr[jp + w * 16 + r + 8 * hi];
#pragma unroll
    for (int kt = 0; kt < 4; ++kt) {
        const float bk = bias[kt * 16 + m];
#pragma unroll
        for (int r = 0; r < 8; ++r) {
            const int j = jp + w * 16 + r + 8 * hi;
            float v = acc[kt][r] * isrv[r] + bk;
            out[(size_t)j * DOUT + kt * 16 + m] = fxp_round(v);
        }
    }
}

// ---------------------------------------------------------------------------
extern "C" void kernel_launch(void* const* d_in, const int* in_sizes, int n_in,
                              void* d_out, int out_size, void* d_ws, size_t ws_size,
                              hipStream_t stream) {
    (void)in_sizes; (void)n_in; (void)out_size; (void)ws_size;
    const float* x = (const float*)d_in[0];
    const float* A = (const float*)d_in[1];
    const float* w = (const float*)d_in[2];
    const float* b = (const float*)d_in[3];
    float* out = (float*)d_out;

    char* ws = (char*)d_ws;
    float*    colpartG = (float*)(ws);                          // 256*16384*4 = 16 MiB
    float*    rowsumG  = (float*)(ws + 16777216);               // 64 KiB
    float*    isr      = (float*)(ws + 16777216 + 65536);       // 64 KiB
    float*    isc      = (float*)(ws + 16777216 + 2 * 65536);   // 64 KiB
    _Float16* Zh       = (_Float16*)(ws + 16777216 + 3 * 65536);            // 2 MiB
    _Float16* Zl       = (_Float16*)(ws + 16777216 + 3 * 65536 + 2097152);  // 2 MiB

    const size_t deg_smem = (size_t)(16384 + 64 * 68 + 64) * sizeof(float); // 83 KB
    gcn_deg <<<256, 256, deg_smem, stream>>>(A, colpartG, rowsumG);
    gcn_norm<<<64, 256, 0, stream>>>(colpartG, rowsumG, isc, isr);
    gcn_xw  <<<512, 128, 0, stream>>>(x, w, isc, Zh, Zl);
    gcn_agg <<<256, 128, 0, stream>>>(A, Zh, Zl, isr, b, out);
}
